// VisionTransformer_74354473828563
// MI455X (gfx1250) — matrix operand order repaired and checked by reading, not compiled, not run on hardware
//
#include <hip/hip_runtime.h>
#include <hip/hip_bf16.h>

// ======================= model dimensions =======================
#define BB    16
#define MMt   4
#define CCc   3
#define IMGQ  224
#define PPp   16
#define DDd   768
#define LLl   12
#define HHh   12
#define DFFf  3072
#define NCc   1000
#define GGg   14
#define SSs   197
#define HDh   64
#define BS_   (BB*SSs)        // 3152
#define BMS_  (BB*MMt*SSs)    // 12608
#define PK_   (CCc*PPp*PPp)   // 768
#define NPATCH (GGg*GGg)      // 196
#define EPSf  1e-6f
#define SCALEf 0.125f         // 64^-0.5

typedef __attribute__((ext_vector_type(16))) __bf16 v16bf;
typedef __attribute__((ext_vector_type(8)))  __bf16 v8bf;
typedef __attribute__((ext_vector_type(2)))  __bf16 v2bf;
typedef __attribute__((ext_vector_type(8)))  float  v8f;

// pack two f32 -> two bf16 in one u32 (single v_cvt_pk_bf16_f32)
static __device__ inline unsigned pk_bf16(float a, float b) {
    v2bf v;
    v[0] = (__bf16)a;
    v[1] = (__bf16)b;
    return __builtin_bit_cast(unsigned, v);
}

// ======================= WMMA GEMM =======================
// C[M,N] = alpha * A[M,K] * op(B) (+ bias[N]) (+ C)
//   bkn==0 : B is (N,K) row-major  (C = A * B^T)
//   bkn==1 : B is (K,N) row-major  (C = A * B)
// Batch z = (b*nm + m)*nh + h with per-operand strides.
// 2-stage software pipeline: next K-tile is fetched into registers while the
// current tile's WMMAs run; regs -> LDS conversion happens after the barrier.
#define TBM 128
#define TBN 64
#define TBK 32

__global__ __launch_bounds__(256)
void gemm_bf16_wmma(const float* __restrict__ A, const float* __restrict__ Bm,
                    const float* __restrict__ bias, float* __restrict__ C,
                    int Md, int Nd, int Kd,
                    long lda, long ldb, long ldc,
                    int nm, int nh,
                    long sAb, long sAm, long sAh,
                    long sBb, long sBm, long sBh,
                    long sCb, long sCm, long sCh,
                    int bkn, float alpha, int accum, int addb)
{
    int z  = blockIdx.z;
    int zh = z % nh;  int zt = z / nh;
    int zm = zt % nm; int zb = zt / nm;
    A  += (long)zb*sAb + (long)zm*sAm + (long)zh*sAh;
    Bm += (long)zb*sBb + (long)zm*sBm + (long)zh*sBh;
    C  += (long)zb*sCb + (long)zm*sCm + (long)zh*sCh;

    const int row0 = blockIdx.y * TBM;
    const int col0 = blockIdx.x * TBN;
    const bool fullM = (row0 + TBM <= Md);
    const bool fullN = (col0 + TBN <= Nd);

    __shared__ __align__(16) unsigned AsU[TBM*TBK/2];
    __shared__ __align__(16) unsigned BsU[TBN*TBK/2];
    const __bf16* As = (const __bf16*)AsU;
    const __bf16* Bs = (const __bf16*)BsU;

    const int tid  = threadIdx.x;
    const int lane = tid & 31;
    const int w    = tid >> 5;          // 8 waves
    const int wm   = w & 3;             // 4 waves along M (32 rows each)
    const int wn   = w >> 2;            // 2 waves along N (32 cols each)
    const int half = lane >> 4;         // 0: lanes0-15, 1: lanes16-31
    const int lr   = lane & 15;

    // staging registers for the in-flight tile
    float ar[16];
    float br[8];

    auto load_tile = [&](int kb) {
        const bool fullK = (kb + TBK <= Kd);
        // ---- A tile -> ar ----
        if (fullM & fullK) {
#pragma unroll
            for (int i = 0; i < 8; ++i) {
                int e = tid + i*256;            // pair index, 2048 pairs
                int m = e >> 4, kp = e & 15;
                long base = (long)(row0 + m)*lda + kb + 2*kp;
                ar[2*i]   = A[base];
                ar[2*i+1] = A[base + 1];
            }
        } else {
#pragma unroll
            for (int i = 0; i < 8; ++i) {
                int e = tid + i*256;
                int m = e >> 4, kp = e & 15;
                int gm = row0 + m;
                int gk0 = kb + 2*kp, gk1 = gk0 + 1;
                int cm  = gm  < Md ? gm  : Md - 1;
                int ck0 = gk0 < Kd ? gk0 : Kd - 1;
                int ck1 = gk1 < Kd ? gk1 : Kd - 1;
                float f0 = A[(long)cm*lda + ck0];
                float f1 = A[(long)cm*lda + ck1];
                ar[2*i]   = (gm < Md && gk0 < Kd) ? f0 : 0.0f;
                ar[2*i+1] = (gm < Md && gk1 < Kd) ? f1 : 0.0f;
            }
        }
        // ---- B tile -> br ----
        if (!bkn) {
            if (fullN & fullK) {
#pragma unroll
                for (int i = 0; i < 4; ++i) {
                    int e = tid + i*256;        // 1024 pairs
                    int n = e >> 4, kp = e & 15;
                    long base = (long)(col0 + n)*ldb + kb + 2*kp;
                    br[2*i]   = Bm[base];
                    br[2*i+1] = Bm[base + 1];
                }
            } else {
#pragma unroll
                for (int i = 0; i < 4; ++i) {
                    int e = tid + i*256;
                    int n = e >> 4, kp = e & 15;
                    int gn = col0 + n;
                    int gk0 = kb + 2*kp, gk1 = gk0 + 1;
                    int cn  = gn  < Nd ? gn  : Nd - 1;
                    int ck0 = gk0 < Kd ? gk0 : Kd - 1;
                    int ck1 = gk1 < Kd ? gk1 : Kd - 1;
                    float f0 = Bm[(long)cn*ldb + ck0];
                    float f1 = Bm[(long)cn*ldb + ck1];
                    br[2*i]   = (gn < Nd && gk0 < Kd) ? f0 : 0.0f;
                    br[2*i+1] = (gn < Nd && gk1 < Kd) ? f1 : 0.0f;
                }
            }
        } else {
            if (fullN & fullK) {
#pragma unroll
                for (int i = 0; i < 4; ++i) {
                    int e = tid + i*256;
                    int n = e & 63, kp = e >> 6;   // n contiguous across threads
                    long base = (long)(kb + 2*kp)*ldb + col0 + n;
                    br[2*i]   = Bm[base];
                    br[2*i+1] = Bm[base + ldb];
                }
            } else {
#pragma unroll
                for (int i = 0; i < 4; ++i) {
                    int e = tid + i*256;
                    int n = e & 63, kp = e >> 6;
                    int gn = col0 + n;
                    int gk0 = kb + 2*kp, gk1 = gk0 + 1;
                    int cn  = gn  < Nd ? gn  : Nd - 1;
                    int ck0 = gk0 < Kd ? gk0 : Kd - 1;
                    int ck1 = gk1 < Kd ? gk1 : Kd - 1;
                    float f0 = Bm[(long)ck0*ldb + cn];
                    float f1 = Bm[(long)ck1*ldb + cn];
                    br[2*i]   = (gn < Nd && gk0 < Kd) ? f0 : 0.0f;
                    br[2*i+1] = (gn < Nd && gk1 < Kd) ? f1 : 0.0f;
                }
            }
        }
    };

    auto store_tile = [&]() {
#pragma unroll
        for (int i = 0; i < 8; ++i) {
            int e = tid + i*256;
            int m = e >> 4, kp = e & 15;
            AsU[m*16 + kp] = pk_bf16(ar[2*i], ar[2*i+1]);
        }
        if (!bkn) {
#pragma unroll
            for (int i = 0; i < 4; ++i) {
                int e = tid + i*256;
                int n = e >> 4, kp = e & 15;
                BsU[n*16 + kp] = pk_bf16(br[2*i], br[2*i+1]);
            }
        } else {
#pragma unroll
            for (int i = 0; i < 4; ++i) {
                int e = tid + i*256;
                int n = e & 63, kp = e >> 6;
                BsU[n*16 + kp] = pk_bf16(br[2*i], br[2*i+1]);
            }
        }
    };

    v8f acc[2][2];
#pragma unroll
    for (int i = 0; i < 2; ++i)
#pragma unroll
        for (int j = 0; j < 2; ++j)
#pragma unroll
            for (int r = 0; r < 8; ++r) acc[i][j][r] = 0.0f;

    // pipeline prologue: fetch tile 0
    load_tile(0);

    for (int kb = 0; kb < Kd; kb += TBK) {
        // commit the tile fetched last iteration to LDS
        store_tile();
        __syncthreads();

        // kick off global loads for the NEXT tile; they overlap the WMMAs below
        if (kb + TBK < Kd) {
            load_tile(kb + TBK);
            // L2 prefetch two tiles ahead (global_prefetch_b8)
            if (kb + 2*TBK < Kd) {
                int gm = row0 + w*16;
                if (gm < Md && lane == 0)
                    __builtin_prefetch(&A[(long)gm*lda + kb + 2*TBK], 0, 1);
            }
        }

        // ---------- gather WMMA fragments per ISA 7.12.2 layouts ----------
        v16bf afrag[2], bfrag[2];
#pragma unroll
        for (int mt = 0; mt < 2; ++mt) {
            int r = wm*32 + mt*16 + lr;
            const __bf16* p = &As[r*TBK + half*8];   // lanes0-15: K 0-7 / lanes16-31: K 8-15
            v8bf lo = *(const v8bf*)p;
            v8bf hi = *(const v8bf*)(p + 16);        // K 16-23 / 24-31
#pragma unroll
            for (int i = 0; i < 8; ++i) { afrag[mt][i] = lo[i]; afrag[mt][i+8] = hi[i]; }
        }
#pragma unroll
        for (int nt = 0; nt < 2; ++nt) {
            int c = wn*32 + nt*16 + lr;
            const __bf16* p = &Bs[c*TBK + half*8];   // lanes0-15: K 0-7 / lanes16-31: K 8-15
            v8bf lo = *(const v8bf*)p;
            v8bf hi = *(const v8bf*)(p + 16);        // K 16-23 / 24-31
#pragma unroll
            for (int i = 0; i < 8; ++i) { bfrag[nt][i] = lo[i]; bfrag[nt][i+8] = hi[i]; }
        }
#pragma unroll
        for (int mt = 0; mt < 2; ++mt)
#pragma unroll
            for (int nt = 0; nt < 2; ++nt)
                acc[mt][nt] = __builtin_amdgcn_wmma_f32_16x16x32_bf16(
                    false, afrag[mt], false, bfrag[nt],
                    (short)0, acc[mt][nt], false, false);
        __syncthreads();   // LDS free for next iteration's store_tile
    }

    // ---------- epilogue: C/D layout (VGPR r -> M base / base+8; lane halves) ----------
    if (fullM & fullN) {
#pragma unroll
        for (int mt = 0; mt < 2; ++mt)
#pragma unroll
            for (int nt = 0; nt < 2; ++nt) {
                int gn = col0 + wn*32 + nt*16 + lr;
                float bv = addb ? bias[gn] : 0.0f;
#pragma unroll
                for (int r = 0; r < 8; ++r) {
                    int gm = row0 + wm*32 + mt*16 + half*8 + r;
                    long ci = (long)gm*ldc + gn;
                    float v = acc[mt][nt][r] * alpha + bv;
                    if (accum) C[ci] += v; else C[ci] = v;
                }
            }
    } else {
#pragma unroll
        for (int mt = 0; mt < 2; ++mt)
#pragma unroll
            for (int nt = 0; nt < 2; ++nt)
#pragma unroll
                for (int r = 0; r < 8; ++r) {
                    int gm = row0 + wm*32 + mt*16 + half*8 + r;
                    int gn = col0 + wn*32 + nt*16 + lr;
                    if (gm < Md && gn < Nd) {
                        float v = acc[mt][nt][r] * alpha;
                        if (addb) v += bias[gn];
                        long ci = (long)gm*ldc + gn;
                        if (accum) C[ci] += v; else C[ci] = v;
                    }
                }
    }
}

// ======================= reductions =======================
__device__ inline float block_sum(float v, float* red) {
    int tid = threadIdx.x;
    red[tid] = v; __syncthreads();
    for (int s = 128; s > 0; s >>= 1) { if (tid < s) red[tid] += red[tid+s]; __syncthreads(); }
    float r = red[0]; __syncthreads();
    return r;
}
__device__ inline float block_max(float v, float* red) {
    int tid = threadIdx.x;
    red[tid] = v; __syncthreads();
    for (int s = 128; s > 0; s >>= 1) { if (tid < s) red[tid] = fmaxf(red[tid], red[tid+s]); __syncthreads(); }
    float r = red[0]; __syncthreads();
    return r;
}

// ======================= LayerNorm JVP =======================
__global__ __launch_bounds__(256)
void ln_primal_kernel(const float* __restrict__ x, const float* __restrict__ g,
                      const float* __restrict__ bt, float* __restrict__ y,
                      float* __restrict__ xh, float* __restrict__ inv)
{
    long r = blockIdx.x;
    const float* xr = x + r*(long)DDd;
    __shared__ float red[256];
    int tid = threadIdx.x;
    float v[3];
    float s0 = 0.f;
#pragma unroll
    for (int i = 0; i < 3; ++i) { v[i] = xr[tid + i*256]; s0 += v[i]; }
    float mu = block_sum(s0, red) * (1.0f/DDd);
    float s1 = 0.f;
#pragma unroll
    for (int i = 0; i < 3; ++i) { v[i] -= mu; s1 += v[i]*v[i]; }
    float var = block_sum(s1, red) * (1.0f/DDd);
    float iv = rsqrtf(var + EPSf);
#pragma unroll
    for (int i = 0; i < 3; ++i) {
        int d = tid + i*256;
        float h = v[i]*iv;
        xh[r*(long)DDd + d] = h;
        y [r*(long)DDd + d] = h*g[d] + bt[d];
    }
    if (tid == 0) inv[r] = iv;
}

__global__ __launch_bounds__(256)
void ln_tangent_kernel(const float* __restrict__ t, const float* __restrict__ xh,
                       const float* __restrict__ inv, const float* __restrict__ g,
                       float* __restrict__ ty)
{
    long r = blockIdx.x;                     // row in (B,M,S)
    long b = r / (MMt*SSs);
    int  s = (int)(r % SSs);
    long pr = b*SSs + s;                     // primal row
    const float* tr = t  + r*(long)DDd;
    const float* xr = xh + pr*(long)DDd;
    float iv = inv[pr];
    __shared__ float red[256];
    int tid = threadIdx.x;
    float tv[3], xv[3];
    float s0 = 0.f, s1 = 0.f;
#pragma unroll
    for (int i = 0; i < 3; ++i) {
        int d = tid + i*256;
        tv[i] = tr[d]; xv[i] = xr[d];
        s0 += tv[i]; s1 += xv[i]*tv[i];
    }
    float mt  = block_sum(s0, red) * (1.0f/DDd);
    float mxt = block_sum(s1, red) * (1.0f/DDd);
#pragma unroll
    for (int i = 0; i < 3; ++i) {
        int d = tid + i*256;
        ty[r*(long)DDd + d] = g[d]*iv*(tv[i] - mt - xv[i]*mxt);
    }
}

// ======================= Softmax JVP =======================
__global__ __launch_bounds__(256)
void softmax_kernel(float* __restrict__ sm)           // one block per (b,h,q) row, in place
{
    long r = blockIdx.x;
    float* row = sm + r*(long)SSs;
    __shared__ float red[256];
    int tid = threadIdx.x;
    float mx = -3.4e38f;
    for (int c = tid; c < SSs; c += 256) mx = fmaxf(mx, row[c]);
    mx = block_max(mx, red);
    float acc = 0.f;
    for (int c = tid; c < SSs; c += 256) { float e = expf(row[c]-mx); row[c] = e; acc += e; }
    float tot = block_sum(acc, red);
    float rinv = 1.0f / tot;
    for (int c = tid; c < SSs; c += 256) row[c] *= rinv;
}

__global__ __launch_bounds__(256)
void softmax_jvp_kernel(const float* __restrict__ a, float* __restrict__ ts) // one block per (b,m,h,q)
{
    long r = blockIdx.x;
    int q = (int)(r % SSs);  long t1 = r / SSs;
    int h = (int)(t1 % HHh); long t2 = t1 / HHh;
    long b  = t2 / MMt;
    const float* ar = a + (((b*HHh + h)*(long)SSs + q) * (long)SSs);
    float* tr = ts + r*(long)SSs;
    __shared__ float red[256];
    int tid = threadIdx.x;
    float acc = 0.f;
    for (int c = tid; c < SSs; c += 256) acc += ar[c]*tr[c];
    float dot = block_sum(acc, red);
    for (int c = tid; c < SSs; c += 256) tr[c] = ar[c]*(tr[c] - dot);
}

// ======================= GELU JVP =======================
__global__ __launch_bounds__(256)
void gelu_t_kernel(const float* __restrict__ x, float* __restrict__ t, long n)
{
    long e = (long)blockIdx.x*256 + threadIdx.x;
    if (e >= n) return;
    int d = (int)(e % DFFf); long rs = e / DFFf;
    int s = (int)(rs % SSs); long bm = rs / SSs;
    long b = bm / MMt;
    float xv = x[(b*SSs + s)*(long)DFFf + d];
    float cdf = 0.5f*(1.0f + erff(xv*0.70710678118654752f));
    float pdf = expf(-0.5f*xv*xv)*0.3989422804014327f;
    t[e] = t[e]*(cdf + xv*pdf);
}
__global__ __launch_bounds__(256)
void gelu_kernel(float* __restrict__ x, long n)
{
    long e = (long)blockIdx.x*256 + threadIdx.x;
    if (e >= n) return;
    float xv = x[e];
    x[e] = xv * 0.5f*(1.0f + erff(xv*0.70710678118654752f));
}

// ======================= patchify / embed helpers =======================
__global__ __launch_bounds__(256)
void patchify_kernel(const float* __restrict__ img, float* __restrict__ out, long nrows)
{
    long e = (long)blockIdx.x*256 + threadIdx.x;
    long n = nrows * (long)PK_;
    if (e >= n) return;
    long row = e / PK_; int col = (int)(e % PK_);
    long bb = row / NPATCH; int pidx = (int)(row % NPATCH);
    int gy = pidx / GGg, gx = pidx % GGg;
    int c  = col / (PPp*PPp), r2 = col % (PPp*PPp);
    int py = r2 / PPp, px = r2 % PPp;
    long src = ((bb*CCc + c)*(long)IMGQ + (gy*PPp+py))*(long)IMGQ + (gx*PPp+px);
    out[e] = img[src];
}

__global__ __launch_bounds__(256)
void pos_cls_kernel(float* __restrict__ h, const float* __restrict__ cls,
                    const float* __restrict__ pos)
{
    long e = (long)blockIdx.x*256 + threadIdx.x;
    long n = (long)BS_*DDd;
    if (e >= n) return;
    int d = (int)(e % DDd); long t = e / DDd;
    int s = (int)(t % SSs);
    if (s == 0) h[e] = cls[d] + pos[d];
    else        h[e] += pos[(long)s*DDd + d];
}

__global__ __launch_bounds__(256)
void zero_tcls_kernel(float* __restrict__ th)
{
    long e = (long)blockIdx.x*256 + threadIdx.x;
    long n = (long)BB*MMt*DDd;
    if (e >= n) return;
    int d = (int)(e % DDd); long bm = e / DDd;
    th[(bm*(long)SSs)*DDd + d] = 0.0f;       // tangent cls row = 0
}

__global__ __launch_bounds__(256)
void gather0_kernel(const float* __restrict__ y, float* __restrict__ y0, long nrows)
{
    long e = (long)blockIdx.x*256 + threadIdx.x;
    long n = nrows*(long)DDd;
    if (e >= n) return;
    int d = (int)(e % DDd); long r = e / DDd;
    y0[e] = y[(r*(long)SSs)*DDd + d];
}

// ======================= host side =======================
static inline long cdiv(long a, long b) { return (a + b - 1) / b; }

static void gemm(hipStream_t st, const float* A, const float* Bm, const float* bias, float* C,
                 int Md, int Nd, int Kd, long lda, long ldb, long ldc,
                 int nb, int nm, int nh,
                 long sAb, long sAm, long sAh,
                 long sBb, long sBm, long sBh,
                 long sCb, long sCm, long sCh,
                 int bkn, float alpha, int accum, int addb)
{
    dim3 grid((unsigned)cdiv(Nd, TBN), (unsigned)cdiv(Md, TBM), (unsigned)(nb*nm*nh));
    gemm_bf16_wmma<<<grid, 256, 0, st>>>(A, Bm, bias, C, Md, Nd, Kd, lda, ldb, ldc,
                                         nm, nh, sAb, sAm, sAh, sBb, sBm, sBh,
                                         sCb, sCm, sCh, bkn, alpha, accum, addb);
}

extern "C" void kernel_launch(void* const* d_in, const int* in_sizes, int n_in,
                              void* d_out, int out_size, void* d_ws, size_t ws_size,
                              hipStream_t stream)
{
    const float* x       = (const float*)d_in[0];
    const float* xt      = (const float*)d_in[1];
    const float* patch_w = (const float*)d_in[2];
    const float* patch_b = (const float*)d_in[3];
    const float* cls_tok = (const float*)d_in[4];
    const float* pos_emb = (const float*)d_in[5];
    const float* ln1_g   = (const float*)d_in[6];
    const float* ln1_b   = (const float*)d_in[7];
    const float* qkv_w   = (const float*)d_in[8];
    const float* qkv_b   = (const float*)d_in[9];
    const float* proj_w  = (const float*)d_in[10];
    const float* proj_b  = (const float*)d_in[11];
    const float* ln2_g   = (const float*)d_in[12];
    const float* ln2_b   = (const float*)d_in[13];
    const float* fc1_w   = (const float*)d_in[14];
    const float* fc1_b   = (const float*)d_in[15];
    const float* fc2_w   = (const float*)d_in[16];
    const float* fc2_b   = (const float*)d_in[17];
    const float* normf_g = (const float*)d_in[18];
    const float* normf_b = (const float*)d_in[19];
    const float* head_w  = (const float*)d_in[20];
    const float* head_b  = (const float*)d_in[21];
    float* out = (float*)d_out;

    // ---- workspace bump layout (floats), ~643 MB total ----
    float* ws = (float*)d_ws;
    long off = 0;
    float* h    = ws + off; off += (long)BS_*DDd;
    float* th   = ws + off; off += (long)BMS_*DDd;
    float* y    = ws + off; off += (long)BS_*DDd;
    float* ty   = ws + off; off += (long)BMS_*DDd;
    float* xh   = ws + off; off += (long)BS_*DDd;
    float* inv  = ws + off; off += 4096;
    float* qkv  = ws + off; off += (long)BS_*3*DDd;
    float* tqkv = ws + off; off += (long)BMS_*3*DDd;
    float* attn = ws + off; off += (long)BB*HHh*SSs*SSs;
    float* tatt = ws + off; off += (long)BB*MMt*HHh*SSs*SSs;
    float* o    = ws + off; off += (long)BS_*DDd;
    float* to   = ws + off; off += (long)BMS_*DDd;
    float* mid  = ws + off; off += (long)BS_*DFFf;
    float* tmid = ws + off; off += (long)BMS_*DFFf;
    float* y0   = ws + off; off += (long)BB*DDd;
    float* ty0  = ws + off; off += (long)BB*MMt*DDd;
    // patch staging aliases tmid region (dead at this point in the schedule)
    float* apat  = tmid;
    float* apatt = tmid + (long)BB*NPATCH*PK_;

    const long Z = 0;

    // ============ patch embedding (primal + tangent) ============
    {
        long n1 = (long)BB*NPATCH*PK_;
        patchify_kernel<<<(unsigned)cdiv(n1,256), 256, 0, stream>>>(x, apat, (long)BB*NPATCH);
        long n2 = (long)BB*MMt*NPATCH*PK_;
        patchify_kernel<<<(unsigned)cdiv(n2,256), 256, 0, stream>>>(xt, apatt, (long)BB*MMt*NPATCH);
        gemm(stream, apat, patch_w, patch_b, h + DDd,
             NPATCH, DDd, PK_, PK_, PK_, DDd,
             BB, 1, 1,
             (long)NPATCH*PK_, Z, Z,  Z, Z, Z,  (long)SSs*DDd, Z, Z,
             0, 1.0f, 0, 1);
        gemm(stream, apatt, patch_w, nullptr, th + DDd,
             NPATCH, DDd, PK_, PK_, PK_, DDd,
             BB*MMt, 1, 1,
             (long)NPATCH*PK_, Z, Z,  Z, Z, Z,  (long)SSs*DDd, Z, Z,
             0, 1.0f, 0, 0);
        long n3 = (long)BS_*DDd;
        pos_cls_kernel<<<(unsigned)cdiv(n3,256), 256, 0, stream>>>(h, cls_tok, pos_emb);
        long n4 = (long)BB*MMt*DDd;
        zero_tcls_kernel<<<(unsigned)cdiv(n4,256), 256, 0, stream>>>(th);
    }

    const long sQKV = (long)SSs*3*DDd;      // per-b row-block stride in qkv
    const long sAtt = (long)SSs*SSs;

    // ============ transformer layers ============
    for (int l = 0; l < LLl; ++l) {
        const float* g1 = ln1_g + l*DDd;  const float* b1 = ln1_b + l*DDd;
        const float* g2 = ln2_g + l*DDd;  const float* b2 = ln2_b + l*DDd;
        const float* qw = qkv_w + (long)l*3*DDd*DDd;  const float* qb = qkv_b + (long)l*3*DDd;
        const float* pw = proj_w + (long)l*DDd*DDd;   const float* pb = proj_b + (long)l*DDd;
        const float* w1 = fc1_w + (long)l*DFFf*DDd;   const float* c1 = fc1_b + (long)l*DFFf;
        const float* w2 = fc2_w + (long)l*DDd*DFFf;   const float* c2 = fc2_b + (long)l*DDd;

        // LN1 (saves xh, inv for the tangent pass)
        ln_primal_kernel<<<BS_, 256, 0, stream>>>(h, g1, b1, y, xh, inv);
        ln_tangent_kernel<<<BMS_, 256, 0, stream>>>(th, xh, inv, g1, ty);

        // qkv / tqkv
        gemm(stream, y,  qw, qb, qkv,  BS_,  3*DDd, DDd, DDd, DDd, 3*DDd,
             1,1,1, Z,Z,Z, Z,Z,Z, Z,Z,Z, 0, 1.0f, 0, 1);
        gemm(stream, ty, qw, nullptr, tqkv, BMS_, 3*DDd, DDd, DDd, DDd, 3*DDd,
             1,1,1, Z,Z,Z, Z,Z,Z, Z,Z,Z, 0, 1.0f, 0, 0);

        // s = SCALE * Q K^T   (batched over b,h)
        gemm(stream, qkv, qkv + DDd, nullptr, attn,
             SSs, SSs, HDh, 3*DDd, 3*DDd, SSs,
             BB, 1, HHh,
             sQKV, Z, HDh,   sQKV, Z, HDh,   (long)HHh*sAtt, Z, sAtt,
             0, SCALEf, 0, 0);
        // ts = SCALE * (TQ K^T)           (batched over b,m,h)
        gemm(stream, tqkv, qkv + DDd, nullptr, tatt,
             SSs, SSs, HDh, 3*DDd, 3*DDd, SSs,
             BB, MMt, HHh,
             (long)MMt*sQKV, sQKV, HDh,   sQKV, Z, HDh,
             (long)MMt*HHh*sAtt, (long)HHh*sAtt, sAtt,
             0, SCALEf, 0, 0);
        // ts += SCALE * (Q TK^T)
        gemm(stream, qkv, tqkv + DDd, nullptr, tatt,
             SSs, SSs, HDh, 3*DDd, 3*DDd, SSs,
             BB, MMt, HHh,
             sQKV, Z, HDh,   (long)MMt*sQKV, sQKV, HDh,
             (long)MMt*HHh*sAtt, (long)HHh*sAtt, sAtt,
             0, SCALEf, 1, 0);

        // softmax + softmax-JVP (in place)
        softmax_kernel<<<(unsigned)((long)BB*HHh*SSs), 256, 0, stream>>>(attn);
        softmax_jvp_kernel<<<(unsigned)((long)BB*MMt*HHh*SSs), 256, 0, stream>>>(attn, tatt);

        // o = A V    (B is (K,N): V rows stride 3*D)
        gemm(stream, attn, qkv + 2*DDd, nullptr, o,
             SSs, HDh, SSs, SSs, 3*DDd, DDd,
             BB, 1, HHh,
             (long)HHh*sAtt, Z, sAtt,   sQKV, Z, HDh,   (long)SSs*DDd, Z, HDh,
             1, 1.0f, 0, 0);
        // to = TA V
        gemm(stream, tatt, qkv + 2*DDd, nullptr, to,
             SSs, HDh, SSs, SSs, 3*DDd, DDd,
             BB, MMt, HHh,
             (long)MMt*HHh*sAtt, (long)HHh*sAtt, sAtt,   sQKV, Z, HDh,
             (long)MMt*SSs*DDd, (long)SSs*DDd, HDh,
             1, 1.0f, 0, 0);
        // to += A TV
        gemm(stream, attn, tqkv + 2*DDd, nullptr, to,
             SSs, HDh, SSs, SSs, 3*DDd, DDd,
             BB, MMt, HHh,
             (long)HHh*sAtt, Z, sAtt,   (long)MMt*sQKV, sQKV, HDh,
             (long)MMt*SSs*DDd, (long)SSs*DDd, HDh,
             1, 1.0f, 1, 0);

        // residual: h += o @ proj^T + proj_b ; th += to @ proj^T
        gemm(stream, o,  pw, pb,     h,  BS_,  DDd, DDd, DDd, DDd, DDd,
             1,1,1, Z,Z,Z, Z,Z,Z, Z,Z,Z, 0, 1.0f, 1, 1);
        gemm(stream, to, pw, nullptr, th, BMS_, DDd, DDd, DDd, DDd, DDd,
             1,1,1, Z,Z,Z, Z,Z,Z, Z,Z,Z, 0, 1.0f, 1, 0);

        // LN2
        ln_primal_kernel<<<BS_, 256, 0, stream>>>(h, g2, b2, y, xh, inv);
        ln_tangent_kernel<<<BMS_, 256, 0, stream>>>(th, xh, inv, g2, ty);

        // MLP fc1
        gemm(stream, y,  w1, c1,     mid,  BS_,  DFFf, DDd, DDd, DDd, DFFf,
             1,1,1, Z,Z,Z, Z,Z,Z, Z,Z,Z, 0, 1.0f, 0, 1);
        gemm(stream, ty, w1, nullptr, tmid, BMS_, DFFf, DDd, DDd, DDd, DFFf,
             1,1,1, Z,Z,Z, Z,Z,Z, Z,Z,Z, 0, 1.0f, 0, 0);

        // GELU JVP (tangent reads pre-activation, so run it first; then primal in place)
        long nt = (long)BMS_*DFFf;
        gelu_t_kernel<<<(unsigned)cdiv(nt,256), 256, 0, stream>>>(mid, tmid, nt);
        long np = (long)BS_*DFFf;
        gelu_kernel<<<(unsigned)cdiv(np,256), 256, 0, stream>>>(mid, np);

        // MLP fc2 (+ residual)
        gemm(stream, mid,  w2, c2,     h,  BS_,  DDd, DFFf, DFFf, DFFf, DDd,
             1,1,1, Z,Z,Z, Z,Z,Z, Z,Z,Z, 0, 1.0f, 1, 1);
        gemm(stream, tmid, w2, nullptr, th, BMS_, DDd, DFFf, DFFf, DFFf, DDd,
             1,1,1, Z,Z,Z, Z,Z,Z, Z,Z,Z, 0, 1.0f, 1, 0);
    }

    // ============ final norm + head ============
    ln_primal_kernel<<<BS_, 256, 0, stream>>>(h, normf_g, normf_b, y, xh, inv);
    ln_tangent_kernel<<<BMS_, 256, 0, stream>>>(th, xh, inv, normf_g, ty);

    long ng = (long)BB*DDd;
    gather0_kernel<<<(unsigned)cdiv(ng,256), 256, 0, stream>>>(y, y0, BB);
    long ngt = (long)BB*MMt*DDd;
    gather0_kernel<<<(unsigned)cdiv(ngt,256), 256, 0, stream>>>(ty, ty0, (long)BB*MMt);

    // logits = y0 @ head^T + head_b ; tlogits = ty0 @ head^T
    gemm(stream, y0,  head_w, head_b, out,        BB,      NCc, DDd, DDd, DDd, NCc,
         1,1,1, Z,Z,Z, Z,Z,Z, Z,Z,Z, 0, 1.0f, 0, 1);
    gemm(stream, ty0, head_w, nullptr, out + (long)BB*NCc, BB*MMt, NCc, DDd, DDd, DDd, NCc,
         1,1,1, Z,Z,Z, Z,Z,Z, Z,Z,Z, 0, 1.0f, 0, 0);
}
